// pairGraph_5514738008861
// MI455X (gfx1250) — compile-verified, matrix operand order checked by reading, not run on hardware
//
#include <hip/hip_runtime.h>
#include <hip/hip_bf16.h>
#include <math.h>

// ---------------------------------------------------------------------------
// Types for CDNA5 WMMA
// ---------------------------------------------------------------------------
typedef __attribute__((ext_vector_type(16))) _Float16 v16h;
typedef __attribute__((ext_vector_type(8)))  _Float16 v8h;
typedef __attribute__((ext_vector_type(8)))  float    v8f;

// ---------------------------------------------------------------------------
// Tiled WMMA GEMM:  C[M,N] = act(A[M,K] @ W[K,N] + bias)
//   - A, W fp32 row-major in global; converted to f16 in LDS
//   - 256 threads = 8 wave32; block tile 128x128, K-step 32
//   - wave (wm,wn) in 4x2 grid computes 32x64 via 2x4 v_wmma_f32_16x16x32_f16
//   - Requires M%128==0, N%128==0, K%32==0 (true for all call sites here)
// act: 0 = none, 1 = relu
// ---------------------------------------------------------------------------
#define LDS_STRIDE 40   // 32 halfs + 8 pad => 80B rows (16B-aligned)

__global__ __launch_bounds__(256)
void gemm_wmma_f16(const float* __restrict__ A, const float* __restrict__ Wt,
                   const float* __restrict__ bias, float* __restrict__ C,
                   int M, int K, int Nout, int act) {
  __shared__ _Float16 sA[128 * LDS_STRIDE];
  __shared__ _Float16 sB[128 * LDS_STRIDE];

  const int tid  = threadIdx.x;
  const int wave = tid >> 5;
  const int lane = tid & 31;
  const int wm   = wave >> 1;          // 0..3 : 32-row strip
  const int wn   = wave & 1;           // 0..1 : 64-col strip
  const int lr   = lane & 15;          // row (A) / col (B) within 16
  const int lh   = (lane >> 4) * 8;    // K-half base per ISA layout
  const int bM   = blockIdx.y * 128;
  const int bN   = blockIdx.x * 128;

  v8f acc[2][4] = {};

  const int lr128 = tid >> 1;          // 0..127
  const int seg   = (tid & 1) * 16;    // 0 or 16

  for (int kk = 0; kk < K; kk += 32) {
    // ---- stage A tile: 128 rows x 32 k, fp32 -> f16, row-major ----
    {
      const float* g = A + (size_t)(bM + lr128) * K + kk + seg;
      _Float16* s = &sA[lr128 * LDS_STRIDE + seg];
#pragma unroll
      for (int j = 0; j < 16; j += 4) {
        float4 f = *(const float4*)(g + j);
        s[j + 0] = (_Float16)f.x; s[j + 1] = (_Float16)f.y;
        s[j + 2] = (_Float16)f.z; s[j + 3] = (_Float16)f.w;
      }
      if (kk + 32 < K)  // emits global_prefetch_b8
        __builtin_prefetch(g + 32, 0, 0);
    }
    // ---- stage B tile transposed: sB[n][k], n=0..127, k=0..31 ----
    {
      const float* g = Wt + (size_t)(kk + seg) * Nout + bN + lr128;
      _Float16* s = &sB[lr128 * LDS_STRIDE + seg];
#pragma unroll
      for (int j = 0; j < 16; ++j)
        s[j] = (_Float16)g[(size_t)j * Nout];
    }
    __syncthreads();

    // ---- build fragments per ISA 16-bit A/B layout:
    //      lane needs halfs [lh..lh+7] and [lh+16..lh+23] of its row/col ----
    v16h af[2], bf[4];
#pragma unroll
    for (int i = 0; i < 2; ++i) {
      const _Float16* p = &sA[(wm * 32 + i * 16 + lr) * LDS_STRIDE + lh];
      v8h lo = *(const v8h*)p;
      v8h hi = *(const v8h*)(p + 16);
      af[i] = __builtin_shufflevector(lo, hi, 0,1,2,3,4,5,6,7,8,9,10,11,12,13,14,15);
    }
#pragma unroll
    for (int j = 0; j < 4; ++j) {
      const _Float16* p = &sB[(wn * 64 + j * 16 + lr) * LDS_STRIDE + lh];
      v8h lo = *(const v8h*)p;
      v8h hi = *(const v8h*)(p + 16);
      bf[j] = __builtin_shufflevector(lo, hi, 0,1,2,3,4,5,6,7,8,9,10,11,12,13,14,15);
    }

#pragma unroll
    for (int i = 0; i < 2; ++i)
#pragma unroll
      for (int j = 0; j < 4; ++j)
        acc[i][j] = __builtin_amdgcn_wmma_f32_16x16x32_f16(
            false, af[i], false, bf[j], (short)0, acc[i][j], false, false);
    __syncthreads();
  }

  // ---- epilogue: C/D layout — lane n = lane&15, m = v + (lane>>4)*8 ----
  const int mrow0 = bM + wm * 32;
  const int ncol0 = bN + wn * 64;
  const int mhalf = (lane >> 4) * 8;
#pragma unroll
  for (int i = 0; i < 2; ++i)
#pragma unroll
    for (int j = 0; j < 4; ++j) {
      const int n = ncol0 + j * 16 + lr;
      const float bv = bias ? bias[n] : 0.0f;
#pragma unroll
      for (int v = 0; v < 8; ++v) {
        const int m = mrow0 + i * 16 + mhalf + v;
        float val = acc[i][j][v] + bv;
        if (act) val = val > 0.0f ? val : 0.0f;
        C[(size_t)m * Nout + n] = val;
      }
    }
}

// ---------------------------------------------------------------------------
// Skinny GEMM (K=8 bbox projection): plain VALU, relu
// ---------------------------------------------------------------------------
__global__ void k_small_gemm_relu(const float* __restrict__ A, const float* __restrict__ Wt,
                                  const float* __restrict__ b, float* __restrict__ C,
                                  int M, int K, int Nout) {
  int idx = blockIdx.x * blockDim.x + threadIdx.x;
  if (idx >= M * Nout) return;
  int m = idx / Nout, n = idx - m * Nout;
  float s = b[n];
  for (int k = 0; k < K; ++k) s += A[(size_t)m * K + k] * Wt[(size_t)k * Nout + n];
  C[idx] = s > 0.0f ? s : 0.0f;
}

// concat(personA, personB) -> [B,1024]
__global__ void k_concat(const float* __restrict__ a, const float* __restrict__ b,
                         float* __restrict__ out, int B) {
  int idx = blockIdx.x * blockDim.x + threadIdx.x;
  if (idx >= B * 1024) return;
  int i = idx >> 10, c = idx & 1023;
  out[idx] = (c < 512) ? a[i * 512 + c] : b[i * 512 + c - 512];
}

// serial exclusive scan over img_rel_num (B=1024, trivial)
__global__ void k_scan(const int* __restrict__ rel, int* __restrict__ csum, int B) {
  if (blockIdx.x == 0 && threadIdx.x == 0) {
    int s = 0;
    for (int i = 0; i < B; ++i) { csum[i] = s; s += rel[i]; }
  }
}

// build src[] (graph pos -> stacked row) and pair_pos[]
__global__ void k_fill_idx(const int* __restrict__ rel, const int* __restrict__ csum,
                           int* __restrict__ src, int* __restrict__ pair_pos,
                           int B, int Np) {
  int i = blockIdx.x * blockDim.x + threadIdx.x;
  if (i >= B) return;
  int base = csum[i], r = rel[i];
  for (int j = 0; j < r; ++j) {
    int g = base + i + j;          // pair graph position
    src[g] = base + j;
    pair_pos[base + j] = g;
  }
  src[base + r + i] = Np + i;      // person node after its image's pairs
}

// x[g] = stacked[src[g]]  (stacked = [personPair ; fc_person]), float4 rows
__global__ void k_gather(const int* __restrict__ src, const float4* __restrict__ pp,
                         const float4* __restrict__ fcp, float4* __restrict__ x,
                         int N, int Np) {
  int idx = blockIdx.x * blockDim.x + threadIdx.x;
  if (idx >= N * 128) return;
  int g = idx >> 7, c = idx & 127;
  int s = src[g];
  x[idx] = (s < Np) ? pp[(size_t)s * 128 + c] : fcp[(size_t)(s - Np) * 128 + c];
}

// segment-sum: agg[d[e]] += m[s[e]], one block per edge, 4 cols/thread
__global__ void k_edge_scatter(const int* __restrict__ ei, const float* __restrict__ m,
                               float* __restrict__ agg, int E) {
  int e = blockIdx.x;
  int c = threadIdx.x * 4;
  int s = ei[e], d = ei[E + e];
  float4 v = *(const float4*)(m + (size_t)s * 512 + c);
  float* o = agg + (size_t)d * 512 + c;
  atomicAdd(o + 0, v.x); atomicAdd(o + 1, v.y);
  atomicAdd(o + 2, v.z); atomicAdd(o + 3, v.w);
}

// GRU cell elementwise; gi/gh hold [rows,1536] with biases already added
__global__ void k_gru(const float* __restrict__ gi, const float* __restrict__ gh,
                      float* __restrict__ x, int rows) {
  int idx = blockIdx.x * blockDim.x + threadIdx.x;
  if (idx >= rows * 512) return;
  int r = idx >> 9, c = idx & 511;
  const float* gir = gi + (size_t)r * 1536;
  const float* ghr = gh + (size_t)r * 1536;
  float rr = 1.0f / (1.0f + __expf(-(gir[c] + ghr[c])));
  float zz = 1.0f / (1.0f + __expf(-(gir[512 + c] + ghr[512 + c])));
  float nn = tanhf(gir[1024 + c] + rr * ghr[1024 + c]);
  float h  = x[idx];
  x[idx] = (1.0f - zz) * nn + zz * h;
}

// result_filter = sigmoid(x[pair_pos])
__global__ void k_sigfilter(const float* __restrict__ x, const int* __restrict__ pair_pos,
                            float* __restrict__ rf, int Np) {
  int idx = blockIdx.x * blockDim.x + threadIdx.x;
  if (idx >= Np * 512) return;
  int p = idx >> 9, c = idx & 511;
  float v = x[(size_t)pair_pos[p] * 512 + c];
  rf[idx] = 1.0f / (1.0f + __expf(-v));
}

// fc_pairClass: N=6, skinny -> VALU
__global__ void k_cls(const float* __restrict__ rf, const float* __restrict__ Wt,
                      const float* __restrict__ b, float* __restrict__ out, int Np) {
  int idx = blockIdx.x * blockDim.x + threadIdx.x;
  if (idx >= Np * 6) return;
  int p = idx / 6, j = idx - p * 6;
  const float* r = rf + (size_t)p * 512;
  float s = b[j];
  for (int c = 0; c < 512; ++c) s += r[c] * Wt[c * 6 + j];
  out[idx] = s;
}

// ---------------------------------------------------------------------------
// Host orchestration
// ---------------------------------------------------------------------------
extern "C" void kernel_launch(void* const* d_in, const int* in_sizes, int n_in,
                              void* d_out, int out_size, void* d_ws, size_t ws_size,
                              hipStream_t stream) {
  (void)n_in; (void)out_size; (void)ws_size;
  const float* pair      = (const float*)d_in[0];
  const float* person_a  = (const float*)d_in[1];
  const float* person_b  = (const float*)d_in[2];
  const float* bbox      = (const float*)d_in[3];
  const int*   rel       = (const int*)d_in[4];
  const int*   edge_idx  = (const int*)d_in[5];
  const float* W_pair    = (const float*)d_in[6];
  const float* b_pair    = (const float*)d_in[7];
  const float* W_pa      = (const float*)d_in[8];
  const float* b_pa      = (const float*)d_in[9];
  const float* W_pb      = (const float*)d_in[10];
  const float* b_pb      = (const float*)d_in[11];
  const float* W_bbox    = (const float*)d_in[12];
  const float* b_bbox    = (const float*)d_in[13];
  const float* W_fcp     = (const float*)d_in[14];
  const float* b_fcp     = (const float*)d_in[15];
  const float* ggc_w     = (const float*)d_in[16];
  const float* gru_W_ih  = (const float*)d_in[17];
  const float* gru_W_hh  = (const float*)d_in[18];
  const float* gru_b_ih  = (const float*)d_in[19];
  const float* gru_b_hh  = (const float*)d_in[20];
  const float* W_cls     = (const float*)d_in[21];
  const float* b_cls     = (const float*)d_in[22];

  const int SZ = 512, IN = 1024;
  const int Np = in_sizes[0] / IN;       // 32256
  const int B  = in_sizes[1] / IN;       // 1024
  const int N  = Np + B;                 // 33280
  const int E  = in_sizes[5] / 2;        // 133120

  // output layout (return-order, fp32)
  float* out   = (float*)d_out;
  float* o_cls = out;
  float* o_pp  = o_cls + (size_t)Np * 6;
  float* o_pa  = o_pp  + (size_t)Np * SZ;
  float* o_pb  = o_pa  + (size_t)B  * SZ;
  float* o_h   = o_pb  + (size_t)B  * SZ;
  float* o_rf  = o_h   + (size_t)Np * SZ;

  // workspace carve (256B aligned)
  char* wp = (char*)d_ws;
  auto carve = [&](size_t bytes) -> void* {
    void* p = (void*)wp; wp += (bytes + 255) & ~(size_t)255; return p;
  };
  int*   ws_csum = (int*)carve((size_t)B * 4);
  int*   ws_src  = (int*)carve((size_t)N * 4);
  int*   ws_pp   = (int*)carve((size_t)Np * 4);
  float* ws_cat  = (float*)carve((size_t)B * 1024 * 4);
  float* ws_fcp  = (float*)carve((size_t)B * SZ * 4);
  float* ws_x    = (float*)carve((size_t)N * SZ * 4);
  float* ws_m    = (float*)carve((size_t)N * SZ * 4);
  float* ws_agg  = (float*)carve((size_t)N * SZ * 4);
  const int tilesM    = N / 128;                 // 260
  const int chunkRows = ((tilesM + 4) / 5) * 128; // 6656
  float* ws_gi = (float*)carve((size_t)chunkRows * 1536 * 4);
  float* ws_gh = (float*)carve((size_t)chunkRows * 1536 * 4);

  auto gemm = [&](const float* A, const float* Wm, const float* bias, float* C,
                  int M, int K, int Nout, int act) {
    dim3 grid(Nout / 128, M / 128);
    gemm_wmma_f16<<<grid, dim3(256), 0, stream>>>(A, Wm, bias, C, M, K, Nout, act);
  };

  // --- projections (WMMA) ---
  gemm(pair,     W_pair, b_pair, o_pp, Np, IN, SZ, 1);
  gemm(person_a, W_pa,   b_pa,   o_pa, B,  IN, SZ, 1);
  gemm(person_b, W_pb,   b_pb,   o_pb, B,  IN, SZ, 1);
  k_small_gemm_relu<<<(Np * SZ + 255) / 256, 256, 0, stream>>>(bbox, W_bbox, b_bbox, o_h, Np, 8, SZ);

  // --- fc_person ---
  k_concat<<<(B * 1024 + 255) / 256, 256, 0, stream>>>(o_pa, o_pb, ws_cat, B);
  gemm(ws_cat, W_fcp, b_fcp, ws_fcp, B, 2 * SZ, SZ, 0);

  // --- ragged graph indices + gather ---
  k_scan<<<1, 1, 0, stream>>>(rel, ws_csum, B);
  k_fill_idx<<<(B + 255) / 256, 256, 0, stream>>>(rel, ws_csum, ws_src, ws_pp, B, Np);
  k_gather<<<(N * 128 + 255) / 256, 256, 0, stream>>>(ws_src, (const float4*)o_pp,
                                                      (const float4*)ws_fcp, (float4*)ws_x, N, Np);

  // --- GatedGraphConv x3 ---
  for (int layer = 0; layer < 3; ++layer) {
    gemm(ws_x, ggc_w + (size_t)layer * SZ * SZ, nullptr, ws_m, N, SZ, SZ, 0);
    hipMemsetAsync(ws_agg, 0, (size_t)N * SZ * 4, stream);
    k_edge_scatter<<<E, 128, 0, stream>>>(edge_idx, ws_m, ws_agg, E);
    for (int r0 = 0; r0 < N; r0 += chunkRows) {
      int rows = (N - r0 < chunkRows) ? (N - r0) : chunkRows;
      gemm(ws_agg + (size_t)r0 * SZ, gru_W_ih, gru_b_ih, ws_gi, rows, SZ, 3 * SZ, 0);
      gemm(ws_x   + (size_t)r0 * SZ, gru_W_hh, gru_b_hh, ws_gh, rows, SZ, 3 * SZ, 0);
      k_gru<<<(rows * SZ + 255) / 256, 256, 0, stream>>>(ws_gi, ws_gh, ws_x + (size_t)r0 * SZ, rows);
    }
  }

  // --- sigmoid, filter, classify ---
  k_sigfilter<<<(Np * SZ + 255) / 256, 256, 0, stream>>>(ws_x, ws_pp, o_rf, Np);
  k_cls<<<(Np * 6 + 255) / 256, 256, 0, stream>>>(o_rf, W_cls, b_cls, o_cls, Np);
}
